// Model_57939108823224
// MI455X (gfx1250) — matrix operand order repaired and checked by reading, not compiled, not run on hardware
//
#include <hip/hip_runtime.h>
#include <hip/hip_bf16.h>

// ---------------------------------------------------------------------------
// Fused MoE patch-expert kernel for gfx1250 (MI455X).
//   Phase A: Y = x_blk(32x512) @ We[:, p*512:(p+1)*512] + b_e   (f16 WMMA, f32 acc)
//   Phase B: Z = Y(32x512) @ Wp                                  (f16 WMMA, f32 acc)
//   Epilogue: combined[rb, seq, :] += gate * (Z + bp)            (global f32 atomics)
// ---------------------------------------------------------------------------

typedef __attribute__((ext_vector_type(16))) _Float16 v16h;
typedef __attribute__((ext_vector_type(8)))  _Float16 h8;
typedef __attribute__((ext_vector_type(4)))  _Float16 h4;
typedef __attribute__((ext_vector_type(8)))  float    v8f;

#define D_MODEL  512
#define SEQ      96
#define KSTEP    32
#define MBLK     32
#define NTHREADS 256
#define ASTR     520   // padded f16 stride for A/Y tile (conflict-free b128 reads)
#define BSTR     40    // padded f16 stride for B^T slab
#define BLOCKS_PER_EXPERT 768   // 8*L*P == 8*96

#if defined(__has_builtin)
#  if __has_builtin(__builtin_amdgcn_sched_barrier)
#    define SCHED_FENCE() __builtin_amdgcn_sched_barrier(0)
#  endif
#endif
#ifndef SCHED_FENCE
#  define SCHED_FENCE() asm volatile("" ::: "memory")
#endif

struct MoeParams {
    const float* xs[8];
    const float* W[8];
    const float* b[8];
    const float* gates;
    const float* Wp;
    const float* bp;
};

union Frag { h8 h[2]; v16h v; };

__global__ void zero_f4(float4* __restrict__ out, int n4) {
    int i = blockIdx.x * blockDim.x + threadIdx.x;
    if (i < n4) out[i] = make_float4(0.f, 0.f, 0.f, 0.f);
}

// stage a 32x512 f32 k-slab of B into LDS as B^T f16 [n][k]; float4 global reads
__device__ __forceinline__
void stage_bt(_Float16* __restrict__ BT, const float* __restrict__ src,
              int ldsrc, int nbase, int k0, int tid)
{
    for (int idx4 = tid; idx4 < (KSTEP * D_MODEL) / 4; idx4 += NTHREADS) {
        const int r  = idx4 >> 7;            // k within slab, 0..31
        const int c4 = (idx4 & 127) * 4;     // n, multiple of 4 (coalesced over tid)
        const float4 v = *(const float4*)&src[(size_t)(k0 + r) * ldsrc + nbase + c4];
        BT[(c4 + 0) * BSTR + r] = (_Float16)v.x;
        BT[(c4 + 1) * BSTR + r] = (_Float16)v.y;
        BT[(c4 + 2) * BSTR + r] = (_Float16)v.z;
        BT[(c4 + 3) * BSTR + r] = (_Float16)v.w;
    }
}

// prefetch next k-slab (one 256B-granular prefetch per thread covers 64KB slab)
__device__ __forceinline__
void prefetch_slab(const float* __restrict__ src, int ldsrc, int nbase, int kn, int tid)
{
    const int r  = tid >> 3;           // 0..31
    const int cc = (tid & 7) * 64;     // 8 chunks of 64 floats per row
    __builtin_prefetch(&src[(size_t)(kn + r) * ldsrc + nbase + cc], 0, 1);
}

// load A fragment + all 8 B fragments, fence the scheduler, then issue the
// 8 WMMAs back-to-back (XDL pipelining; avoids per-WMMA s_wait_dscnt 0)
__device__ __forceinline__
void mma_step(const _Float16* __restrict__ Alds, const _Float16* __restrict__ BT,
              int k0, int mr0, int nstrip, int lane16, int laneHi, v8f acc[8])
{
    Frag af;
    const int arow = mr0 + lane16;
    af.h[0] = *(const h8*)&Alds[arow * ASTR + k0 + laneHi * 8];
    af.h[1] = *(const h8*)&Alds[arow * ASTR + k0 + 16 + laneHi * 8];

    Frag bf[8];
#pragma unroll
    for (int nt = 0; nt < 8; ++nt) {
        const int c0 = nstrip + nt * 16 + lane16;
        bf[nt].h[0] = *(const h8*)&BT[c0 * BSTR + laneHi * 8];
        bf[nt].h[1] = *(const h8*)&BT[c0 * BSTR + 16 + laneHi * 8];
    }
    SCHED_FENCE();
#pragma unroll
    for (int nt = 0; nt < 8; ++nt) {
        acc[nt] = __builtin_amdgcn_wmma_f32_16x16x32_f16(
            false, af.v, false, bf[nt].v, (short)0, acc[nt], false, false);
    }
}

__global__ __launch_bounds__(NTHREADS)
void moe_fused(MoeParams prm, float* __restrict__ out)
{
    const int Ptab[8] = {2, 4, 6, 8, 12, 16, 24, 48};

    const int blk = blockIdx.x;
    const int e   = blk / BLOCKS_PER_EXPERT;
    const int rem = blk - e * BLOCKS_PER_EXPERT;
    const int P   = Ptab[e];
    const int L   = SEQ / P;
    const int mTiles = BLOCKS_PER_EXPERT / P;     // == 8*L == 256*L/MBLK
    const int mt  = rem % mTiles;                 // consecutive blocks share p-block of We
    const int p   = rem / mTiles;
    const int M0  = mt * MBLK;

    const float* __restrict__ x   = prm.xs[e];
    const float* __restrict__ We  = prm.W[e];
    const float* __restrict__ be  = prm.b[e];
    const int ldW   = P * D_MODEL;
    const int nbase = p * D_MODEL;

    extern __shared__ char smem[];
    _Float16* Alds = (_Float16*)smem;                 // MBLK x ASTR   (x, then Y)
    _Float16* BT   = Alds + MBLK * ASTR;              // D_MODEL x BSTR (B^T slab)

    const int tid    = threadIdx.x;
    const int lane   = tid & 31;
    const int wave   = tid >> 5;
    const int lane16 = lane & 15;
    const int laneHi = lane >> 4;
    const int mr0    = (wave >> 2) * 16;              // 0 or 16
    const int nstrip = (wave & 3) * 128;              // 0,128,256,384

    v8f acc[8];
#pragma unroll
    for (int t = 0; t < 8; ++t) acc[t] = (v8f)(0.0f);

    // ---- stage A = x block, f32 -> f16 (float4 reads, b64 LDS stores) ----
    for (int idx4 = tid; idx4 < (MBLK * D_MODEL) / 4; idx4 += NTHREADS) {
        const int r  = idx4 >> 7;
        const int c4 = (idx4 & 127) * 4;
        const float4 v = *(const float4*)&x[(size_t)(M0 + r) * D_MODEL + c4];
        h4 hv; hv.x = (_Float16)v.x; hv.y = (_Float16)v.y;
               hv.z = (_Float16)v.z; hv.w = (_Float16)v.w;
        *(h4*)&Alds[r * ASTR + c4] = hv;
    }
    __syncthreads();

    // =======================  Phase A: Y = x @ We[:, p-block]  ==============
    for (int ks = 0; ks < D_MODEL / KSTEP; ++ks) {
        const int k0 = ks * KSTEP;
        if (ks) __syncthreads();
        stage_bt(BT, We, ldW, nbase, k0, tid);
        __syncthreads();
        if (ks + 1 < D_MODEL / KSTEP)
            prefetch_slab(We, ldW, nbase, k0 + KSTEP, tid);

        mma_step(Alds, BT, k0, mr0, nstrip, lane16, laneHi, acc);
    }
    __syncthreads();   // all waves done reading x tile

    // ---- Y + bias -> f16 back into Alds ----
#pragma unroll
    for (int nt = 0; nt < 8; ++nt) {
        const int col   = nstrip + nt * 16 + lane16;
        const float bias = be[nbase + col];
#pragma unroll
        for (int g = 0; g < 8; ++g) {
            const int row = mr0 + laneHi * 8 + g;
            Alds[row * ASTR + col] = (_Float16)(acc[nt][g] + bias);
        }
        acc[nt] = (v8f)(0.0f);
    }
    __syncthreads();

    // =========================  Phase B: Z = Y @ Wp  ========================
    for (int ks = 0; ks < D_MODEL / KSTEP; ++ks) {
        const int k0 = ks * KSTEP;
        if (ks) __syncthreads();
        stage_bt(BT, prm.Wp, D_MODEL, 0, k0, tid);
        __syncthreads();
        if (ks + 1 < D_MODEL / KSTEP)
            prefetch_slab(prm.Wp, D_MODEL, 0, k0 + KSTEP, tid);

        mma_step(Alds, BT, k0, mr0, nstrip, lane16, laneHi, acc);
    }

    // ---- Epilogue: closed-form gather inverse + gate + atomic scatter-add ----
    // expert e covers batch rows r with r%8 in {e, (e+7)%8}; stitched order is
    // row-ascending, so sample i -> rb = 8*(i/2) + (i odd ? hi : lo).
    const int r1 = e, r2 = (e + 7) & 7;
    const int lo = r1 < r2 ? r1 : r2;
    const int hi = r1 < r2 ? r2 : r1;

    int   rbArr[8];
    int   seqArr[8];
    float gArr[8];
#pragma unroll
    for (int g = 0; g < 8; ++g) {
        const int m  = M0 + mr0 + laneHi * 8 + g;     // flattened (sample, l)
        const int i  = m / L;
        const int l  = m - i * L;
        const int rb = ((i >> 1) << 3) + ((i & 1) ? hi : lo);
        rbArr[g]  = rb;
        seqArr[g] = l * P + p;
        gArr[g]   = prm.gates[rb * 8 + e];
    }
#pragma unroll
    for (int nt = 0; nt < 8; ++nt) {
        const int col   = nstrip + nt * 16 + lane16;
        const float bpv = prm.bp[col];
#pragma unroll
        for (int g = 0; g < 8; ++g) {
            const float v = gArr[g] * (acc[nt][g] + bpv);
            atomicAdd(&out[((size_t)rbArr[g] * SEQ + seqArr[g]) * D_MODEL + col], v);
        }
    }
}

extern "C" void kernel_launch(void* const* d_in, const int* in_sizes, int n_in,
                              void* d_out, int out_size, void* d_ws, size_t ws_size,
                              hipStream_t stream)
{
    (void)in_sizes; (void)n_in; (void)d_ws; (void)ws_size;

    MoeParams prm;
    for (int i = 0; i < 8; ++i) {
        prm.xs[i] = (const float*)d_in[3 * i + 0];
        prm.W[i]  = (const float*)d_in[3 * i + 1];
        prm.b[i]  = (const float*)d_in[3 * i + 2];
    }
    prm.gates = (const float*)d_in[24];
    prm.Wp    = (const float*)d_in[25];
    prm.bp    = (const float*)d_in[26];

    float* out = (float*)d_out;

    // zero-init output (scatter-add target); out_size = 1024*96*512, /4 exact
    const int n4 = out_size / 4;
    zero_f4<<<(n4 + NTHREADS - 1) / NTHREADS, NTHREADS, 0, stream>>>((float4*)out, n4);

    const size_t smemBytes = (size_t)(MBLK * ASTR + D_MODEL * BSTR) * sizeof(_Float16); // 74,240 B
    moe_fused<<<8 * BLOCKS_PER_EXPERT, NTHREADS, smemBytes, stream>>>(prm, out);
}
